// Linear4bit_9199819948153
// MI455X (gfx1250) — compile-verified, matrix operand order checked by reading, not run on hardware
//
#include <hip/hip_runtime.h>

// ---------------------------------------------------------------------------
// NF4 Linear (bitsandbytes-style) as a bf16 WMMA GEMM for gfx1250 (MI455X).
//   out[m,n] = sum_k x[m,k] * (NF4_CODE[w_idx[n,k]] * absmax[(n*K+k)/64]) + bias[n]
// M = 8192, N = 4096, K = 4096 (assumed multiples of 128/256/64).
//
// Block tile 128(M) x 256(N), K-step 32 (one WMMA K).  256 threads = 8 waves
// in a 2(M) x 4(N) grid; each wave owns 64x64 = 4x4 WMMA tiles (16 wmma/step).
// LDS double-buffered, one barrier per K-step.  Pipeline (steady state,
// branch-free): LDS-store tile t+1, global-load tile t+2, prefetch tile t+3,
// then 16 WMMAs on tile t.  Drain iterations are peeled out of the hot loop.
// ---------------------------------------------------------------------------

typedef __bf16 bf16_t;
typedef __attribute__((ext_vector_type(16))) __bf16 v16bf;
typedef __attribute__((ext_vector_type(8)))  float  v8f;
typedef __attribute__((ext_vector_type(4)))  float  v4f;
typedef __attribute__((ext_vector_type(4)))  int    v4i;

__device__ __constant__ float NF4_CODE_D[16] = {
    -1.0f, -0.6961928009986877f, -0.5250730514526367f, -0.39491748809814453f,
    -0.28444138169288635f, -0.18477343022823334f, -0.09105003625154495f, 0.0f,
    0.07958029955625534f, 0.16093020141124725f, 0.24611230194568634f,
    0.33791524171829224f, 0.44070982933044434f, 0.5626170039176941f,
    0.7229568362236023f, 1.0f};

#define BM   128
#define BN   256
#define BK   32
#define ASTR 40   // LDS row stride in bf16 elems (32 data + 8 pad); 80B = 5*16B

__global__ __launch_bounds__(256)
void nf4_linear_wmma(const float* __restrict__ x,
                     const int*   __restrict__ widx,
                     const float* __restrict__ absmax,
                     const float* __restrict__ bias,
                     float*       __restrict__ out,
                     int M, int N, int K)
{
    __shared__ __align__(16) bf16_t sA[2][BM * ASTR];   // X tiles, bf16, K-major
    __shared__ __align__(16) bf16_t sB[2][BN * ASTR];   // dequantized W tiles
    __shared__ float sCode[16];

    const int tid = threadIdx.x;
    if (tid < 16) sCode[tid] = NF4_CODE_D[tid];

    const int m0 = blockIdx.y * BM;
    const int n0 = blockIdx.x * BN;

    const int wave = tid >> 5;      // 8 waves (wave32)
    const int lane = tid & 31;
    const int wm   = wave >> 2;     // 0..1 : 64-row M slab
    const int wn   = wave & 3;      // 0..3 : 64-col N slab
    const int kh   = lane >> 4;     // K-half selector per ISA 16-bit A/B layout
    const int l16  = lane & 15;     // M (or N) within 16x16 tile

    // --- global->LDS staging: thread t fills X row t/2 and W rows t/2, t/2+128
    const int rowS = tid >> 1;            // 0..127
    const int seg  = (tid & 1) * 16;      // k offset 0 or 16
    const int Kb   = K >> 6;              // absmax blocks per W row

    const float* xrow   = x      + (size_t)(m0 + rowS) * K + seg;
    const int*   wrow0  = widx   + (size_t)(n0 + rowS) * K + seg;
    const int*   wrow1  = widx   + (size_t)(n0 + 128 + rowS) * K + seg;
    const float* amrow0 = absmax + (size_t)(n0 + rowS) * Kb;
    const float* amrow1 = absmax + (size_t)(n0 + 128 + rowS) * Kb;

    v4f   xa[4];
    v4i   wa[2][4];
    float am0 = 0.0f, am1 = 0.0f;

    auto loadRegs = [&](int k0) {
#pragma unroll
        for (int i = 0; i < 4; ++i) {
            xa[i]    = *(const v4f*)(xrow  + k0 + i * 4);
            wa[0][i] = *(const v4i*)(wrow0 + k0 + i * 4);
            wa[1][i] = *(const v4i*)(wrow1 + k0 + i * 4);
        }
        am0 = amrow0[k0 >> 6];
        am1 = amrow1[k0 >> 6];
    };

    auto storeLDS = [&](int buf) {
        // X: fp32 -> bf16
        union { v4i v[2]; bf16_t h[16]; } pa;
#pragma unroll
        for (int i = 0; i < 16; ++i)
            pa.h[i] = (bf16_t)(xa[i >> 2][i & 3]);
        *(v4i*)(&sA[buf][rowS * ASTR + seg])     = pa.v[0];
        *(v4i*)(&sA[buf][rowS * ASTR + seg + 8]) = pa.v[1];

        // W: NF4 dequant -> bf16 (absmax constant across this 32-wide K tile)
#pragma unroll
        for (int r = 0; r < 2; ++r) {
            const float am = r ? am1 : am0;
            union { v4i v[2]; bf16_t h[16]; } pb;
#pragma unroll
            for (int i = 0; i < 16; ++i) {
                const int q = wa[r][i >> 2][i & 3] & 15;
                pb.h[i] = (bf16_t)(sCode[q] * am);
            }
            *(v4i*)(&sB[buf][(r * 128 + rowS) * ASTR + seg])     = pb.v[0];
            *(v4i*)(&sB[buf][(r * 128 + rowS) * ASTR + seg + 8]) = pb.v[1];
        }
    };

    // v16bf fragment per ISA 7.12.2 16-bit A/B layout:
    //  elems 0-7  <- [row][kh*8 .. +7],  elems 8-15 <- [row][16 + kh*8 .. +7]
    auto loadFrag = [&](const bf16_t* s, int row) -> v16bf {
        union { v4i v[2]; v16bf f; } u;
        u.v[0] = *(const v4i*)(s + row * ASTR + kh * 8);
        u.v[1] = *(const v4i*)(s + row * ASTR + 16 + kh * 8);
        return u.f;
    };

    v8f acc[4][4];
#pragma unroll
    for (int mt = 0; mt < 4; ++mt)
#pragma unroll
        for (int nt = 0; nt < 4; ++nt)
            acc[mt][nt] = (v8f)0.0f;

    auto wmmaStep = [&](int p) {
        v16bf afr[4], bfr[4];
#pragma unroll
        for (int mt = 0; mt < 4; ++mt)
            afr[mt] = loadFrag(sA[p], wm * 64 + mt * 16 + l16);
#pragma unroll
        for (int nt = 0; nt < 4; ++nt)
            bfr[nt] = loadFrag(sB[p], wn * 64 + nt * 16 + l16);
#pragma unroll
        for (int mt = 0; mt < 4; ++mt)
#pragma unroll
            for (int nt = 0; nt < 4; ++nt)
                acc[mt][nt] = __builtin_amdgcn_wmma_f32_16x16x32_bf16(
                    false, afr[mt], false, bfr[nt],
                    (short)0, acc[mt][nt], false, false);
    };

    const int NT = K / BK;

    __syncthreads();          // sCode visible before first dequant
    loadRegs(0);
    storeLDS(0);
    if (NT > 1) loadRegs(BK); // regs now hold tile 1
    __syncthreads();

    // ---- steady state: branch-free pipeline body, one barrier per K-step ----
    int t = 0;
    for (; t + 2 < NT; ++t) {
        const int p = t & 1;
        storeLDS(p ^ 1);                    // tile t+1 -> other LDS buffer
        loadRegs((t + 2) * BK);             // tile t+2 -> regs
        // speculative prefetch of tile t+3 (never faults; guard-free)
        __builtin_prefetch(xrow  + (t + 3) * BK, 0, 0);
        __builtin_prefetch(wrow0 + (t + 3) * BK, 0, 0);
        __builtin_prefetch(wrow1 + (t + 3) * BK, 0, 0);
        wmmaStep(p);
        __syncthreads();
    }
    // ---- drain: peeled out of the hot loop ----
    if (NT > 1) {
        const int p = t & 1;
        storeLDS(p ^ 1);                    // last tile -> LDS
        wmmaStep(p);
        __syncthreads();
        ++t;
        wmmaStep(t & 1);                    // final K-step, no barrier needed
    } else {
        wmmaStep(0);
    }

    // Epilogue: C/D layout -> lanes 0-15: M=r, N=lane; lanes 16-31: M=8+r
#pragma unroll
    for (int mt = 0; mt < 4; ++mt) {
        const int gm = m0 + wm * 64 + mt * 16 + kh * 8;
#pragma unroll
        for (int nt = 0; nt < 4; ++nt) {
            const int gn = n0 + wn * 64 + nt * 16 + l16;
            const float bv = bias[gn];
#pragma unroll
            for (int r = 0; r < 8; ++r)
                out[(size_t)(gm + r) * N + gn] = acc[mt][nt][r] + bv;
        }
    }
}

extern "C" void kernel_launch(void* const* d_in, const int* in_sizes, int n_in,
                              void* d_out, int out_size, void* d_ws, size_t ws_size,
                              hipStream_t stream) {
    const float* x      = (const float*)d_in[0];
    const int*   widx   = (const int*)  d_in[1];
    const float* absmax = (const float*)d_in[2];
    const float* bias   = (const float*)d_in[3];
    float*       out    = (float*)d_out;

    const int N = in_sizes[3];              // out_f from bias
    const int K = in_sizes[1] / N;          // in_f from w_idx
    const int M = in_sizes[0] / K;          // batch*seq from x

    dim3 grid(N / BN, M / BM);              // x-dim over N: adjacent blocks share X panel in L2
    nf4_linear_wmma<<<grid, dim3(256), 0, stream>>>(x, widx, absmax, bias, out, M, N, K);
}
